// AdaptiveOutlierLoss_86517821210851
// MI455X (gfx1250) — compile-verified, matrix operand order checked by reading, not run on hardware
//
#include <hip/hip_runtime.h>
#include <math.h>

// ---------------------------------------------------------------------------
// Problem constants (match reference shapes exactly)
// ---------------------------------------------------------------------------
#define B_ROWS 32768
#define C_COLS 2048
#define D_DIM  512

#define BM 128        // rows per workgroup
#define BN 128        // prototype columns per n-tile
#define BK 32         // k-step (WMMA K for bf16)
#define AS_STRIDE 520 // bf16 elements per A row in LDS (1040 B, 16B aligned, bank-spread)
#define BS_STRIDE 40  // bf16 elements per B row in LDS (80 B = 64 B data + 16 B TDM pad)

#define EPSF 1e-7f

typedef __bf16        v16bf __attribute__((ext_vector_type(16)));
typedef float         v8f   __attribute__((ext_vector_type(8)));
typedef float         v4f   __attribute__((ext_vector_type(4)));
typedef unsigned int  v4u   __attribute__((ext_vector_type(4)));
typedef unsigned int  u32x4 __attribute__((ext_vector_type(4)));
typedef int           i32x4 __attribute__((ext_vector_type(4)));
typedef int           i32x8 __attribute__((ext_vector_type(8)));

union FragU { v16bf v; v4u q[2]; };

__device__ __forceinline__ unsigned short f32_to_bf16_bits(float f) {
  unsigned u = __float_as_uint(f);
  u += 0x7FFFu + ((u >> 16) & 1u);           // round-to-nearest-even
  return (unsigned short)(u >> 16);
}
__device__ __forceinline__ float bf16_bits_to_f32(unsigned short s) {
  return __uint_as_float(((unsigned)s) << 16);
}

// ---------------------------------------------------------------------------
// Kernel 1: row norms for z (B) and prototypes (C); also zero the output.
// One wave32 per row.
// ---------------------------------------------------------------------------
__global__ void hyp_norms_kernel(const float* __restrict__ z,
                                 const float* __restrict__ p,
                                 float* __restrict__ x2,
                                 float* __restrict__ y2,
                                 float* __restrict__ out) {
  if (blockIdx.x == 0 && threadIdx.x == 0) out[0] = 0.0f;
  const int lane = threadIdx.x & 31;
  const int gw   = (blockIdx.x * blockDim.x + threadIdx.x) >> 5;
  const int R    = B_ROWS + C_COLS;
  if (gw >= R) return;
  const float* src = (gw < B_ROWS) ? (z + (size_t)gw * D_DIM)
                                   : (p + (size_t)(gw - B_ROWS) * D_DIM);
  float s = 0.0f;
#pragma unroll
  for (int j = 0; j < D_DIM / 32; ++j) {
    float v = src[lane + 32 * j];
    s = fmaf(v, v, s);
  }
#pragma unroll
  for (int m = 16; m >= 1; m >>= 1) s += __shfl_xor(s, m, 32);
  if (lane == 0) {
    if (gw < B_ROWS) x2[gw] = s;
    else             y2[gw - B_ROWS] = s;
  }
}

// ---------------------------------------------------------------------------
// Kernel 2: one-time split of prototypes into bf16 hi/lo planes (row-major,
// 2048 x 512).  Keeps all f32->bf16 VALU work out of the GEMM inner loop and
// makes B tiles pure byte-copies (TDM-eligible).
// ---------------------------------------------------------------------------
__global__ void proto_split_kernel(const float* __restrict__ p,
                                   unsigned short* __restrict__ ph,
                                   unsigned short* __restrict__ pl) {
  const int i = blockIdx.x * blockDim.x + threadIdx.x;   // float4 index
  v4f v = *(const v4f*)(p + (size_t)i * 4);
#pragma unroll
  for (int j = 0; j < 4; ++j) {
    float f = v[j];
    unsigned short hi = f32_to_bf16_bits(f);
    unsigned short lo = f32_to_bf16_bits(f - bf16_bits_to_f32(hi));
    ph[(size_t)i * 4 + j] = hi;
    pl[(size_t)i * 4 + j] = lo;
  }
}

// ---------------------------------------------------------------------------
// TDM descriptor + issue: load one 128x32-bf16 tile (row stride 512 elements
// in global) into LDS with an 80-byte padded row stride (pad 4 DWORDs after
// every 16 DWORDs).  2D tensor -> groups 2/3 zero.
// ---------------------------------------------------------------------------
__device__ __forceinline__ void tdm_load_tile(unsigned lds_off,
                                              const unsigned short* gptr) {
  unsigned long long ga = (unsigned long long)gptr;
  u32x4 g0;
  g0[0] = 1u;                                     // count=1, user mode
  g0[1] = lds_off;                                // lds_addr
  g0[2] = (unsigned)(ga & 0xFFFFFFFFu);           // global_addr[31:0]
  g0[3] = (unsigned)((ga >> 32) & 0x01FFFFFFu)    // global_addr[56:32]
          | (2u << 30);                           // type=2 (image)
  i32x8 g1;
  g1[0] = (int)((1u << 16)      // data_size = 2 bytes
              | (1u << 20)      // pad_enable
              | (3u << 22)      // pad_interval: after 16 DWORDs (64 B)
              | (3u << 25));    // pad_amount: 4 DWORDs (16 B)
  g1[1] = (int)(512u << 16);    // tensor_dim0 = 512 (bits 79:48, low half)
  g1[2] = (int)(2048u << 16);   // tensor_dim1 = 2048 (bits 111:80, low half)
  g1[3] = (int)(32u << 16);     // tile_dim0 = 32 (bits 127:112)
  g1[4] = BN;                   // tile_dim1 = 128 rows (bits 143:128)
  g1[5] = 512;                  // tensor_dim0_stride = 512 (bits 207:160, low)
  g1[6] = 0;
  g1[7] = 0;
  i32x4 z4 = {0, 0, 0, 0};
#if __clang_major__ >= 23
  i32x8 z8 = {0, 0, 0, 0, 0, 0, 0, 0};
  __builtin_amdgcn_tensor_load_to_lds(g0, g1, z4, z4, z8, 0);
#else
  __builtin_amdgcn_tensor_load_to_lds(g0, g1, z4, z4, 0);
#endif
}

// ---------------------------------------------------------------------------
// Kernel 3: fused GEMM (bf16x3 WMMA) + hyperbolic-arg min epilogue + loss.
// Grid: B_ROWS/BM workgroups of 256 threads; 8 waves in a 4(m) x 2(n) grid,
// each wave owns a 32x64 output tile (2 tm x 4 tn WMMA tiles).
// ~308 KB dynamic LDS caps occupancy at 1 workgroup/WGP (2 waves/SIMD);
// amdgpu_waves_per_eu tells the allocator so it can use a large VGPR budget
// instead of spilling accumulators to scratch.
// Dynamic LDS layout (bytes):
//   As_hi [BM][AS_STRIDE] bf16 : 133120  @ 0
//   As_lo [BM][AS_STRIDE] bf16 : 133120  @ 133120
//   Bs hi/lo double buffers    : 4x10240 @ 266240
//   x2s   [BM]            f32  :    512  @ 307200
//   minLds[2*BM]          f32  :   1024  @ 307712
//   sumLds[8]             f32  :     32  @ 308736
// ---------------------------------------------------------------------------
#define OFF_AS_HI  0
#define OFF_AS_LO  133120
#define OFF_BS0_HI 266240
#define OFF_BS0_LO 276480
#define OFF_BS1_HI 286720
#define OFF_BS1_LO 296960
#define OFF_X2S    307200
#define OFF_MIN    307712
#define OFF_SUM    308736
#define SMEM_BYTES 308768

__global__ __launch_bounds__(256)
__attribute__((amdgpu_waves_per_eu(1, 2)))
void hyp_loss_main_kernel(const float* __restrict__ z,
                          const unsigned short* __restrict__ ph,
                          const unsigned short* __restrict__ pl,
                          const float* __restrict__ margin_p,
                          const float* __restrict__ x2,
                          const float* __restrict__ y2,
                          float* __restrict__ out) {
  extern __shared__ char smem[];
  unsigned short* As_hi = (unsigned short*)(smem + OFF_AS_HI);
  unsigned short* As_lo = (unsigned short*)(smem + OFF_AS_LO);
  float*  x2s    = (float*)(smem + OFF_X2S);
  float*  minLds = (float*)(smem + OFF_MIN);
  float*  sumLds = (float*)(smem + OFF_SUM);

  const int tid  = threadIdx.x;
  const int lane = tid & 31;
  const int wid  = tid >> 5;
  const int wm   = wid & 3;   // wave row-group (0..3)  -> rows wm*32..+32
  const int wn   = wid >> 2;  // wave col-group (0..1)  -> cols wn*64..+64
  const int half = lane >> 4; // 0: lanes 0-15, 1: lanes 16-31
  const int l15  = lane & 15;
  const int m0   = blockIdx.x * BM;

  // ---- Phase 0: load full 128x512 A slab, split into bf16 hi/lo in LDS ----
  for (int it = tid; it < BM * (D_DIM / 4); it += 256) {
    const int r  = it >> 7;               // D_DIM/4 == 128 float4 per row
    const int c4 = (it & 127) << 2;
    v4f v = *(const v4f*)(z + (size_t)(m0 + r) * D_DIM + c4);
#pragma unroll
    for (int j = 0; j < 4; ++j) {
      float f = v[j];
      unsigned short hi = f32_to_bf16_bits(f);
      unsigned short lo = f32_to_bf16_bits(f - bf16_bits_to_f32(hi));
      As_hi[r * AS_STRIDE + c4 + j] = hi;
      As_lo[r * AS_STRIDE + c4 + j] = lo;
    }
  }
  if (tid < BM) x2s[tid] = x2[m0 + tid];
  __syncthreads();

  // Per-lane x^2 values for the 16 output rows this lane owns (C/D layout:
  // VGPR i -> M = i (lanes 0-15) or 8+i (lanes 16-31), within tile).
  float x2r[2][8];
#pragma unroll
  for (int tm = 0; tm < 2; ++tm)
#pragma unroll
    for (int i = 0; i < 8; ++i)
      x2r[tm][i] = x2s[wm * 32 + tm * 16 + (half ? (8 + i) : i)];

  float minv[2][8];
#pragma unroll
  for (int tm = 0; tm < 2; ++tm)
#pragma unroll
    for (int i = 0; i < 8; ++i) minv[tm][i] = 3.0e38f;

  const int koff = half ? 8 : 0;  // bf16 A/B lane layout: K 0-7|16-23 vs 8-15|24-31
  const unsigned bs_hi_off[2] = {OFF_BS0_HI, OFF_BS1_HI};
  const unsigned bs_lo_off[2] = {OFF_BS0_LO, OFF_BS1_LO};

  // ---- Main loop over prototype tiles ----
  for (int nt = 0; nt < C_COLS / BN; ++nt) {
    v8f acc[2][4];
#pragma unroll
    for (int tm = 0; tm < 2; ++tm)
#pragma unroll
      for (int tn = 0; tn < 4; ++tn)
        acc[tm][tn] = (v8f){0.f, 0.f, 0.f, 0.f, 0.f, 0.f, 0.f, 0.f};

    const int n0g = nt * BN;
    const size_t tile_base = (size_t)n0g * D_DIM;

    // Prologue: DMA k-tile 0 into buffer 0 (wave 0 drives the TDM).
    if (wid == 0) {
      tdm_load_tile(bs_hi_off[0], ph + tile_base);
      tdm_load_tile(bs_lo_off[0], pl + tile_base);
    }

    for (int kt = 0; kt < D_DIM / BK; ++kt) {
      const int buf = kt & 1;
      if (wid == 0) __builtin_amdgcn_s_wait_tensorcnt(0);
      __syncthreads();  // buffer `buf` ready for all; buffer buf^1 reads done

      // Software pipeline: DMA next k-tile into the other buffer.
      if (kt + 1 < D_DIM / BK && wid == 0) {
        const size_t nb = tile_base + (size_t)(kt + 1) * BK;
        tdm_load_tile(bs_hi_off[buf ^ 1], ph + nb);
        tdm_load_tile(bs_lo_off[buf ^ 1], pl + nb);
      }

      const unsigned short* Bs_hi = (const unsigned short*)(smem + bs_hi_off[buf]);
      const unsigned short* Bs_lo = (const unsigned short*)(smem + bs_lo_off[buf]);

      // A fragments stay live for the whole k-step (two ds_read_b128 each per
      // ISA bf16 layout); B fragments are loaded per-tn to cap live ranges.
      FragU ah[2], al[2];
#pragma unroll
      for (int tm = 0; tm < 2; ++tm) {
        const int idxA = (wm * 32 + tm * 16 + l15) * AS_STRIDE + kt * BK + koff;
        ah[tm].q[0] = *(const v4u*)(As_hi + idxA);
        ah[tm].q[1] = *(const v4u*)(As_hi + idxA + 16);
        al[tm].q[0] = *(const v4u*)(As_lo + idxA);
        al[tm].q[1] = *(const v4u*)(As_lo + idxA + 16);
      }

      // bf16x3 compensated product: hi*hi + hi*lo + lo*hi into f32 acc.
#pragma unroll
      for (int tn = 0; tn < 4; ++tn) {
        const int idxB = (wn * 64 + tn * 16 + l15) * BS_STRIDE + koff;
        FragU bhv, blv;
        bhv.q[0] = *(const v4u*)(Bs_hi + idxB);
        bhv.q[1] = *(const v4u*)(Bs_hi + idxB + 16);
        blv.q[0] = *(const v4u*)(Bs_lo + idxB);
        blv.q[1] = *(const v4u*)(Bs_lo + idxB + 16);
#pragma unroll
        for (int tm = 0; tm < 2; ++tm) {
          acc[tm][tn] = __builtin_amdgcn_wmma_f32_16x16x32_bf16(
              false, ah[tm].v, false, bhv.v, (short)0, acc[tm][tn], false, false);
          acc[tm][tn] = __builtin_amdgcn_wmma_f32_16x16x32_bf16(
              false, ah[tm].v, false, blv.v, (short)0, acc[tm][tn], false, false);
          acc[tm][tn] = __builtin_amdgcn_wmma_f32_16x16x32_bf16(
              false, al[tm].v, false, bhv.v, (short)0, acc[tm][tn], false, false);
        }
      }
    }

    // Epilogue: minimize sq/(1-y^2) per row (monotone in the final arccosh arg).
#pragma unroll
    for (int tn = 0; tn < 4; ++tn) {
      const int c     = n0g + wn * 64 + tn * 16 + l15;
      const float yv  = y2[c];
      const float rby = __builtin_amdgcn_rcpf(fmaxf(1.0f - yv, 1e-6f));
#pragma unroll
      for (int tm = 0; tm < 2; ++tm) {
#pragma unroll
        for (int i = 0; i < 8; ++i) {
          const float xy = acc[tm][tn][i];
          const float sq = fmaxf(x2r[tm][i] + yv - 2.0f * xy, 0.0f);
          minv[tm][i] = fminf(minv[tm][i], sq * rby);
        }
      }
    }
  }

  // ---- Reduce min across the 16 lanes sharing each row ----
#pragma unroll
  for (int tm = 0; tm < 2; ++tm) {
#pragma unroll
    for (int i = 0; i < 8; ++i) {
      float v = minv[tm][i];
      v = fminf(v, __shfl_xor(v, 1, 16));
      v = fminf(v, __shfl_xor(v, 2, 16));
      v = fminf(v, __shfl_xor(v, 4, 16));
      v = fminf(v, __shfl_xor(v, 8, 16));
      minv[tm][i] = v;
    }
  }
  if (l15 == 0) {
#pragma unroll
    for (int tm = 0; tm < 2; ++tm)
#pragma unroll
      for (int i = 0; i < 8; ++i) {
        const int row = wm * 32 + tm * 16 + (half ? (8 + i) : i);
        minLds[wn * BM + row] = minv[tm][i];
      }
  }
  __syncthreads();

  // ---- Final per-row distance + loss, block reduce, atomic accumulate ----
  if (tid < BM) {
    const float mgn = margin_p[0];
    float v  = fminf(minLds[tid], minLds[BM + tid]);
    float ax = fmaxf(1.0f - x2s[tid], 1e-6f);
    float arg = 1.0f + 2.0f * v / ax;
    arg = fmaxf(arg, 1.0f + EPSF);
    float dist = acoshf(arg);
    float term = fmaxf(mgn - dist, 0.0f);
#pragma unroll
    for (int m = 16; m >= 1; m >>= 1) term += __shfl_xor(term, m, 32);
    if ((tid & 31) == 0) sumLds[tid >> 5] = term;
  }
  __syncthreads();
  if (tid == 0) {
    float s = sumLds[0] + sumLds[1] + sumLds[2] + sumLds[3];
    atomicAdd(out, s * (1.0f / (float)B_ROWS));
  }
}

// ---------------------------------------------------------------------------
// Workspace layout (bytes):
//   x2 : 32768 f32  @ 0         (131072)
//   y2 :  2048 f32  @ 131072    (  8192)
//   ph : 1M bf16    @ 139264    (2097152)
//   pl : 1M bf16    @ 2236416   (2097152)
// total ~4.3 MB
// ---------------------------------------------------------------------------
extern "C" void kernel_launch(void* const* d_in, const int* in_sizes, int n_in,
                              void* d_out, int out_size, void* d_ws, size_t ws_size,
                              hipStream_t stream) {
  const float* z      = (const float*)d_in[0];
  const float* p      = (const float*)d_in[1];
  const float* margin = (const float*)d_in[2];
  float* out = (float*)d_out;

  char* ws = (char*)d_ws;
  float* x2 = (float*)ws;
  float* y2 = (float*)(ws + 131072);
  unsigned short* ph = (unsigned short*)(ws + 139264);
  unsigned short* pl = (unsigned short*)(ws + 139264 + 2097152);

  const int rows   = B_ROWS + C_COLS;     // one wave32 per row
  const int blocks = (rows * 32 + 255) / 256;
  hyp_norms_kernel<<<blocks, 256, 0, stream>>>(z, p, x2, y2, out);

  const int pel4 = (C_COLS * D_DIM) / 4;  // float4s in prototypes
  proto_split_kernel<<<pel4 / 256, 256, 0, stream>>>(p, ph, pl);

  hyp_loss_main_kernel<<<B_ROWS / BM, 256, SMEM_BYTES, stream>>>(
      z, ph, pl, margin, x2, y2, out);
}
